// MultiHeadSelfAttention_14199161880966
// MI455X (gfx1250) — compile-verified, matrix operand order checked by reading, not run on hardware
//
#include <hip/hip_runtime.h>

// ---------- types ----------
typedef __attribute__((ext_vector_type(16))) __bf16 v16bf;
typedef __attribute__((ext_vector_type(8)))  __bf16 v8bf;
typedef __attribute__((ext_vector_type(8)))  float  v8f;

union Frag16 { v16bf v; v8bf h[2]; };

// round-to-nearest-even f32 -> bf16 (bit-level)
__device__ __forceinline__ __bf16 to_bf16(float x) {
  union { float f; unsigned u; } a; a.f = x;
  unsigned r = a.u + 0x7FFFu + ((a.u >> 16) & 1u);
  union { unsigned short s; __bf16 b; } o; o.s = (unsigned short)(r >> 16);
  return o.b;
}

// Load one 16x32 bf16 WMMA fragment (A-layout; B symmetric with K-major storage).
// ISA 7.12.2: lane L row (L&15); element e -> K = (e&7) + (e>=8?16:0) + (L>=16?8:0)
__device__ __forceinline__ v16bf load_frag(const __bf16* base, int ld, int lane) {
  int m  = lane & 15;
  int hi = lane >> 4;
  const __bf16* p = base + (size_t)m * ld + hi * 8;
  Frag16 f;
  f.h[0] = *(const v8bf*)(p);
  f.h[1] = *(const v8bf*)(p + 16);
  return f.v;
}

__device__ __forceinline__ v8f wmma_bf16(v16bf a, v16bf b, v8f c) {
  return __builtin_amdgcn_wmma_f32_16x16x32_bf16(false, a, false, b, (short)0, c, false, false);
}

// 32B async global->LDS copy (two B128 ops; INST_OFFSET applies to both addresses).
__device__ __forceinline__ void async_cp32(const void* base, unsigned goff, unsigned ldsoff) {
  asm volatile("global_load_async_to_lds_b128 %0, %1, %2 offset:0"
               :: "v"(ldsoff), "v"(goff), "s"(base) : "memory");
  asm volatile("global_load_async_to_lds_b128 %0, %1, %2 offset:16"
               :: "v"(ldsoff), "v"(goff), "s"(base) : "memory");
}
__device__ __forceinline__ void wait_async0() {
  asm volatile("s_wait_asynccnt 0x0" ::: "memory");
}

// ---------- conversion kernels ----------
__global__ void cvt_f32_bf16_kernel(const float* __restrict__ in, __bf16* __restrict__ out, int n) {
  int i = blockIdx.x * blockDim.x + threadIdx.x;
  if (i < n) out[i] = to_bf16(in[i]);
}

// Wt[n][k] = W[k][n]  (K-major weights so B fragments load contiguously per lane)
__global__ void transpose_w_bf16_kernel(const float* __restrict__ W, __bf16* __restrict__ Wt,
                                        int K, int N) {
  int i = blockIdx.x * blockDim.x + threadIdx.x;
  if (i < K * N) {
    int nn = i / K, kk = i % K;
    Wt[i] = to_bf16(W[(size_t)kk * N + nn]);
  }
}

// ---------- output scatter for QKV ----------
// mode 0/1: [b][h][s][d];  mode 2: V^T [b][h][d][s]
__device__ __forceinline__ size_t qkv_index(int mode, int r, int n) {
  int b = r >> 11, s = r & 2047;   // S = 2048
  int h = n >> 6,  d = n & 63;     // Dh = 64
  if (mode == 2) return ((((size_t)b * 16 + h) * 64) + d) * 2048 + s;
  return ((((size_t)b * 16 + h) * 2048) + s) * 64 + d;
}

// ---------- tiled GEMM: C[M,N] = A[M,K](bf16) * Wt[N,K]^T(bf16) + bias ----------
// 128x128 block tile, k-slices of 32 double-buffered in LDS via async loads.
// 8 waves/block, each computes 64x32. LDS rows padded to 56 elems (112B = 7*16B):
// per-row bank offset 28, gcd(28,64)=4 -> 16 distinct banks across the 16 fragment rows.
#define TS 56

__global__ __launch_bounds__(256) void gemm_bf16_kernel(
    const __bf16* __restrict__ A, const __bf16* __restrict__ Wt,
    const float* __restrict__ bias, void* __restrict__ out,
    int M, int N, int K, int mode) {
  __shared__ __align__(16) __bf16 smem[2][2][128 * TS];   // [buf][A/B][row*TS] = 56 KiB

  int tid  = threadIdx.x;
  int lane = tid & 31;
  int w    = tid >> 5;
  int nbx  = N >> 7;
  int blkM = (blockIdx.x / nbx) * 128;
  int blkN = (blockIdx.x % nbx) * 128;
  int wm = (w >> 2) * 64;   // wave M offset within block tile
  int wn = (w & 3) * 32;    // wave N offset within block tile

  // cooperative staging: each of 256 threads moves 32B of the A tile and 32B of the B tile
  int crow  = tid >> 1;          // 0..127
  int chalf = (tid & 1) * 32;    // byte offset within the 64B row payload
  unsigned ldsA0 = (unsigned)(uintptr_t)&smem[0][0][0] + (unsigned)crow * (TS * 2) + chalf;
  unsigned ldsB0 = (unsigned)(uintptr_t)&smem[0][1][0] + (unsigned)crow * (TS * 2) + chalf;
  unsigned bufStride = (unsigned)((uintptr_t)&smem[1][0][0] - (uintptr_t)&smem[0][0][0]);
  unsigned gA = (unsigned)(blkM + crow) * (unsigned)K * 2u + (unsigned)chalf;
  unsigned gB = (unsigned)(blkN + crow) * (unsigned)K * 2u + (unsigned)chalf;

  v8f acc[4][2] = {};
  int nk = K >> 5;

  // prologue: stage k-tile 0 into buffer 0
  async_cp32(A,  gA, ldsA0);
  async_cp32(Wt, gB, ldsB0);

  for (int kt = 0; kt < nk; ++kt) {
    wait_async0();        // my staging loads complete
    __syncthreads();      // everyone's staging loads complete; prev reads of other buf done

    if (kt + 1 < nk) {
      unsigned bo = ((kt + 1) & 1) ? bufStride : 0u;
      unsigned go = (unsigned)(kt + 1) * 64u;   // 32 elems * 2B
      async_cp32(A,  gA + go, ldsA0 + bo);
      async_cp32(Wt, gB + go, ldsB0 + bo);
    }

    const __bf16* tA = &smem[kt & 1][0][0];
    const __bf16* tB = &smem[kt & 1][1][0];

    // issue ALL fragment loads first (one DS clause, incremental dscnt waits),
    // then drain with the 8 WMMAs — keeps multiple LDS loads in flight.
    v16bf b0 = load_frag(tB + (wn +  0) * TS, TS, lane);
    v16bf b1 = load_frag(tB + (wn + 16) * TS, TS, lane);
    v16bf af[4];
    #pragma unroll
    for (int i = 0; i < 4; ++i)
      af[i] = load_frag(tA + (wm + i * 16) * TS, TS, lane);

    #pragma unroll
    for (int i = 0; i < 4; ++i) {
      acc[i][0] = wmma_bf16(af[i], b0, acc[i][0]);
      acc[i][1] = wmma_bf16(af[i], b1, acc[i][1]);
    }
  }

  int hi = lane >> 4, ln = lane & 15;
  #pragma unroll
  for (int i = 0; i < 4; ++i)
    #pragma unroll
    for (int j = 0; j < 2; ++j) {
      int ncol = blkN + wn + j * 16 + ln;
      float bv = bias ? bias[ncol] : 0.0f;
      #pragma unroll
      for (int v = 0; v < 8; ++v) {
        int r = blkM + wm + i * 16 + v + hi * 8;
        float val = acc[i][j][v] + bv;
        if (mode == 3) ((float*)out)[(size_t)r * N + ncol] = val;
        else           ((__bf16*)out)[qkv_index(mode, r, ncol)] = to_bf16(val);
      }
    }
}

// ---------- flash attention: one wave per (b,h,16-query tile) ----------
__global__ __launch_bounds__(256) void attention_kernel(
    const __bf16* __restrict__ Q, const __bf16* __restrict__ Kc,
    const __bf16* __restrict__ Vt, __bf16* __restrict__ Ob) {
  __shared__ __align__(16) __bf16 pbuf[8][16 * 32];   // per-wave P staging (C->A layout)
  int lane  = threadIdx.x & 31;
  int wslot = threadIdx.x >> 5;
  int wave  = blockIdx.x * 8 + wslot;
  int qt = wave & 127;            // S/16 = 128
  int h  = (wave >> 7) & 15;
  int b  = wave >> 11;

  const __bf16* qbase = Q  + (((size_t)b * 16 + h) * 2048 + (size_t)qt * 16) * 64;
  const __bf16* kbase = Kc + (((size_t)b * 16 + h) * 2048) * 64;
  const __bf16* vbase = Vt + (((size_t)b * 16 + h) * 64) * 2048;

  v16bf qf0 = load_frag(qbase +  0, 64, lane);
  v16bf qf1 = load_frag(qbase + 32, 64, lane);

  float mi[8], li[8];
  v8f acc[4] = {};
  #pragma unroll
  for (int v = 0; v < 8; ++v) { mi[v] = -3.0e38f; li[v] = 0.0f; }

  int hi = lane >> 4, ln = lane & 15;
  __bf16* prow = &pbuf[wslot][0];

  for (int kb = 0; kb < 2048; kb += 32) {
    v8f s0 = {}, s1 = {};
    {
      v16bf k00 = load_frag(kbase + (size_t)kb        * 64 +  0, 64, lane);
      v16bf k10 = load_frag(kbase + (size_t)(kb + 16) * 64 +  0, 64, lane);
      v16bf k01 = load_frag(kbase + (size_t)kb        * 64 + 32, 64, lane);
      v16bf k11 = load_frag(kbase + (size_t)(kb + 16) * 64 + 32, 64, lane);
      s0 = wmma_bf16(qf0, k00, s0);
      s1 = wmma_bf16(qf0, k10, s1);
      s0 = wmma_bf16(qf1, k01, s0);
      s1 = wmma_bf16(qf1, k11, s1);
    }

    #pragma unroll
    for (int v = 0; v < 8; ++v) {
      float x0 = s0[v] * 0.125f, x1 = s1[v] * 0.125f;   // 1/sqrt(64)
      float rm = fmaxf(x0, x1);
      #pragma unroll
      for (int msk = 1; msk < 16; msk <<= 1) rm = fmaxf(rm, __shfl_xor(rm, msk, 32));
      float mnew  = fmaxf(mi[v], rm);
      float alpha = __expf(mi[v] - mnew);
      float p0 = __expf(x0 - mnew), p1 = __expf(x1 - mnew);
      float rs = p0 + p1;
      #pragma unroll
      for (int msk = 1; msk < 16; msk <<= 1) rs += __shfl_xor(rs, msk, 32);
      li[v] = li[v] * alpha + rs;
      mi[v] = mnew;
      #pragma unroll
      for (int dt = 0; dt < 4; ++dt) acc[dt][v] *= alpha;
      int m = v + hi * 8;
      prow[m * 32 + ln]      = to_bf16(p0);
      prow[m * 32 + 16 + ln] = to_bf16(p1);
    }

    __builtin_amdgcn_wave_barrier();   // same-wave DS in-order; block reordering

    v16bf pf;
    {
      Frag16 f;
      const __bf16* p = prow + ln * 32 + hi * 8;
      f.h[0] = *(const v8bf*)(p);
      f.h[1] = *(const v8bf*)(p + 16);
      pf = f.v;
    }

    #pragma unroll
    for (int dt = 0; dt < 4; ++dt) {
      v16bf vf = load_frag(vbase + (size_t)(dt * 16) * 2048 + kb, 2048, lane);
      acc[dt] = wmma_bf16(pf, vf, acc[dt]);
    }
  }

  #pragma unroll
  for (int v = 0; v < 8; ++v) {
    float inv = 1.0f / li[v];
    int q = qt * 16 + v + hi * 8;
    size_t rbase = (((size_t)b * 2048 + q) * 1024) + (size_t)h * 64;
    #pragma unroll
    for (int dt = 0; dt < 4; ++dt)
      Ob[rbase + dt * 16 + ln] = to_bf16(acc[dt][v] * inv);
  }
}

// ---------- host ----------
extern "C" void kernel_launch(void* const* d_in, const int* in_sizes, int n_in,
                              void* d_out, int out_size, void* d_ws, size_t ws_size,
                              hipStream_t stream) {
  const float* X  = (const float*)d_in[0];
  const float* Wq = (const float*)d_in[1];
  const float* bq = (const float*)d_in[2];
  const float* Wk = (const float*)d_in[3];
  const float* bk = (const float*)d_in[4];
  const float* Wv = (const float*)d_in[5];
  const float* bv = (const float*)d_in[6];
  const float* Wo = (const float*)d_in[7];
  const float* bo = (const float*)d_in[8];

  const int M = 4096, N = 1024, K = 1024;     // M = B*S
  const size_t SZ_X = (size_t)M * N * 2;      // 8 MiB
  const size_t SZ_W = (size_t)K * N * 2;      // 2 MiB

  char* ws = (char*)d_ws;                     // total 48 MiB
  __bf16* Xbf = (__bf16*)ws; ws += SZ_X;
  __bf16* WqT = (__bf16*)ws; ws += SZ_W;
  __bf16* WkT = (__bf16*)ws; ws += SZ_W;
  __bf16* WvT = (__bf16*)ws; ws += SZ_W;
  __bf16* WoT = (__bf16*)ws; ws += SZ_W;
  __bf16* Qb  = (__bf16*)ws; ws += SZ_X;
  __bf16* Kb  = (__bf16*)ws; ws += SZ_X;
  __bf16* Vtb = (__bf16*)ws; ws += SZ_X;
  __bf16* Ab  = (__bf16*)ws; ws += SZ_X;

  cvt_f32_bf16_kernel<<<(M * N + 255) / 256, 256, 0, stream>>>(X, Xbf, M * N);
  transpose_w_bf16_kernel<<<(K * N + 255) / 256, 256, 0, stream>>>(Wq, WqT, K, N);
  transpose_w_bf16_kernel<<<(K * N + 255) / 256, 256, 0, stream>>>(Wk, WkT, K, N);
  transpose_w_bf16_kernel<<<(K * N + 255) / 256, 256, 0, stream>>>(Wv, WvT, K, N);
  transpose_w_bf16_kernel<<<(K * N + 255) / 256, 256, 0, stream>>>(Wo, WoT, K, N);

  // (M/128)*(N/128) = 32*8 = 256 blocks, 8 waves each
  gemm_bf16_kernel<<<256, 256, 0, stream>>>(Xbf, WqT, bq, Qb,  M, N, K, 0);
  gemm_bf16_kernel<<<256, 256, 0, stream>>>(Xbf, WkT, bk, Kb,  M, N, K, 1);
  gemm_bf16_kernel<<<256, 256, 0, stream>>>(Xbf, WvT, bv, Vtb, M, N, K, 2);

  attention_kernel<<<512, 256, 0, stream>>>(Qb, Kb, Vtb, Ab);

  gemm_bf16_kernel<<<256, 256, 0, stream>>>(Ab, WoT, bo, d_out, M, N, K, 3);
}
